// Multi_Factor_Attention_86870008529231
// MI455X (gfx1250) — matrix operand order repaired and checked by reading, not compiled, not run on hardware
//
#include <hip/hip_runtime.h>
#include <hip/hip_bf16.h>

// Problem constants (match reference)
#define MF   4
#define DIN  1024
#define DOUT 1024
#define BN   32
#define SN   1024

// Main WMMA kernel tiling
#define ROWS 64            // s-rows per block
#define LDA  1032          // bf16 elements per LDS A-row (1024 + 8 pad -> bank stride 4)

typedef __attribute__((ext_vector_type(16))) __bf16 v16bf;
typedef __attribute__((ext_vector_type(8)))  float  v8f;

union Frag { v16bf v; unsigned u[8]; uint4 q[2]; };

__device__ __forceinline__ unsigned pack_bf16(float a, float b) {
  // round-to-nearest-even f32 -> bf16, packed low/high
  unsigned ua = __float_as_uint(a), ub = __float_as_uint(b);
  ua += 0x7FFFu + ((ua >> 16) & 1u);
  ub += 0x7FFFu + ((ub >> 16) & 1u);
  return (ua >> 16) | (ub & 0xFFFF0000u);
}

// ---------------------------------------------------------------------------
// Kernel W: Ws[m][k][o] (f32) -> Wt[m][o][k] (bf16), LDS-tiled 32x32 transpose
// ---------------------------------------------------------------------------
__global__ void k_wt(const float* __restrict__ Ws, unsigned short* __restrict__ Wt) {
  __shared__ float t[32][33];
  const int m  = blockIdx.z;
  const int k0 = blockIdx.x * 32;
  const int o0 = blockIdx.y * 32;
  const int j  = threadIdx.x & 31;   // fast dim
  const int i  = threadIdx.x >> 5;   // 0..7
#pragma unroll
  for (int it = 0; it < 4; ++it) {
    int k = k0 + it * 8 + i;
    t[it * 8 + i][j] = Ws[((size_t)m * DIN + k) * DOUT + o0 + j];  // coalesced in o
  }
  __syncthreads();
#pragma unroll
  for (int it = 0; it < 4; ++it) {
    int o = o0 + it * 8 + i;
    float v = t[j][it * 8 + i];                 // transposed read
    unsigned uv = __float_as_uint(v);
    uv += 0x7FFFu + ((uv >> 16) & 1u);
    Wt[((size_t)m * DOUT + o) * DIN + k0 + j] = (unsigned short)(uv >> 16);  // coalesced in k
  }
}

// ---------------------------------------------------------------------------
// Kernel A: arg[b][o] = tanh([arg1;arg2][b] @ Ww + bw)   (32 x 1024, K=2048)
// ---------------------------------------------------------------------------
__global__ void k_arg(const float* __restrict__ a1, const float* __restrict__ a2,
                      const float* __restrict__ Ww, const float* __restrict__ bw,
                      float* __restrict__ argW) {
  __shared__ float cat[2 * DIN];
  const int b = blockIdx.x, tid = threadIdx.x;
  for (int d = tid; d < DIN; d += 256) {
    cat[d]       = a1[b * DIN + d];
    cat[DIN + d] = a2[b * DIN + d];
  }
  __syncthreads();
  float acc[4];
#pragma unroll
  for (int j = 0; j < 4; ++j) acc[j] = bw[tid + 256 * j];
  for (int d = 0; d < 2 * DIN; ++d) {
    float x = cat[d];
    const float* wr = Ww + (size_t)d * DOUT;
#pragma unroll
    for (int j = 0; j < 4; ++j) acc[j] += x * wr[tid + 256 * j];
  }
#pragma unroll
  for (int j = 0; j < 4; ++j) argW[b * DOUT + tid + 256 * j] = tanhf(acc[j]);
}

// ---------------------------------------------------------------------------
// Kernel B (main): fused  att[b,m,s] = sum_o tanh(enc[b,s,:]·Ws[m][:,o] + bs[m,o]) * arg[b,o]
// bf16 WMMA 16x16x32, f32 accumulate. Block = (s-tile of 64, m, b), 8 waves.
// Wave blocking: 2 row-tiles x 2 o-tiles -> each A and B fragment feeds 2 WMMAs
// (balances LDS bandwidth vs L2 B-panel bandwidth).
// ---------------------------------------------------------------------------
__global__ void __launch_bounds__(256, 1)
k_att(const float* __restrict__ enc, const unsigned short* __restrict__ Wt,
      const float* __restrict__ bs, const float* __restrict__ argW,
      float* __restrict__ att) {
  extern __shared__ char smem[];
  unsigned short* smemA    = (unsigned short*)smem;                 // ROWS*LDA bf16 = 132096 B
  float*          bsL      = (float*)(smem + ROWS * LDA * 2);       // 1024 f32
  float*          argL     = bsL + DOUT;                            // 1024 f32
  float*          sPartial = argL + DOUT;                           // ROWS f32

  const int tid = threadIdx.x;
  const int s0 = blockIdx.x * ROWS;
  const int m  = blockIdx.y;
  const int b  = blockIdx.z;

  // Stage enc tile (ROWS x DIN) f32 -> bf16 in LDS (one pass, reused by all 8 waves)
  const float* encB = enc + ((size_t)b * SN + s0) * DIN;
#pragma unroll 4
  for (int it = 0; it < (ROWS * DIN) / (256 * 4); ++it) {
    int e = (it * 256 + tid) * 4;
    int row = e >> 10, col = e & (DIN - 1);
    float4 f = *(const float4*)(encB + (size_t)row * DIN + col);
    *(uint2*)(smemA + row * LDA + col) = make_uint2(pack_bf16(f.x, f.y), pack_bf16(f.z, f.w));
  }
  for (int k = tid; k < DOUT; k += 256) {
    bsL[k]  = bs[m * DOUT + k];
    argL[k] = argW[b * DOUT + k];
  }
  if (tid < ROWS) sPartial[tid] = 0.f;
  __syncthreads();

  const int w = tid >> 5, lane = tid & 31;
  const int half = lane >> 4, lr = lane & 15;
  const int rp = w & 1;            // row-pair: rows rp*32 .. rp*32+31 (2 row-tiles)
  const int q  = w >> 1;           // o-range: o-tiles q*16 .. q*16+15
  const int rowBase0 = rp * 32;    // row-tile 0
  const int rowBase1 = rp * 32 + 16;  // row-tile 1

  float rowsum[2][8];
#pragma unroll
  for (int r = 0; r < 2; ++r)
#pragma unroll
    for (int i = 0; i < 8; ++i) rowsum[r][i] = 0.f;

  for (int og = 0; og < 8; ++og) {         // 8 groups x 2 o-tiles = 16 o-tiles/wave
    const int otBase = q * 16 + og * 2;
    v8f acc[2][2] = {};                    // [row-tile][o-tile]
    for (int k = 0; k < DIN; k += 32) {
      // --- A fragments (16x32 bf16, contiguous K: k = 16*half + s): lane = M row, VGPR v -> K pair
      Frag a0, a1;
#pragma unroll
      for (int v = 0; v < 8; ++v) {
        int p = half * 16 + 2 * v;
        a0.u[v] = *(const unsigned*)(smemA + (rowBase0 + lr) * LDA + k + p);
        a1.u[v] = *(const unsigned*)(smemA + (rowBase1 + lr) * LDA + k + p);
      }
      if (k + 32 < DIN)   // pull next B panel into near (WGP) cache
        __builtin_prefetch(Wt + ((size_t)m * DOUT + (otBase * 16 + lr)) * DIN + k + 32 + half * 16, 0, 3);
#pragma unroll
      for (int j = 0; j < 2; ++j) {
        // --- B fragment (32x16 bf16): lane = N col, lane-half splits K 0-15/16-31,
        // contiguous 32B per lane from Wt[m][n][k..k+31]
        const int n = (otBase + j) * 16 + lr;
        Frag bf;
        const uint4* bp = (const uint4*)(Wt + ((size_t)m * DOUT + n) * DIN + k + half * 16);
        bf.q[0] = bp[0];
        bf.q[1] = bp[1];
        acc[0][j] = __builtin_amdgcn_wmma_f32_16x16x32_bf16(
            false, a0.v, false, bf.v, (short)0, acc[0][j], false, false);
        acc[1][j] = __builtin_amdgcn_wmma_f32_16x16x32_bf16(
            false, a1.v, false, bf.v, (short)0, acc[1][j], false, false);
      }
    }
    // Epilogue: tanh(+bias) * arg, accumulate per-row partial sums.
    // C layout: VGPR i -> M = rowBase + i + half*8, N = otile*16 + lr
#pragma unroll
    for (int j = 0; j < 2; ++j) {
      const int o = (otBase + j) * 16 + lr;
      const float bsv = bsL[o], agv = argL[o];
#pragma unroll
      for (int r = 0; r < 2; ++r)
#pragma unroll
        for (int i = 0; i < 8; ++i)
          rowsum[r][i] += tanhf(acc[r][j][i] + bsv) * agv;
    }
  }

  // Reduce over the 16 lanes (N) of each half-wave, then combine waves in LDS.
#pragma unroll
  for (int r = 0; r < 2; ++r) {
#pragma unroll
    for (int i = 0; i < 8; ++i) {
      float v = rowsum[r][i];
      v += __shfl_xor(v, 1);
      v += __shfl_xor(v, 2);
      v += __shfl_xor(v, 4);
      v += __shfl_xor(v, 8);
      if (lr == 0) atomicAdd(&sPartial[rp * 32 + r * 16 + half * 8 + i], v);
    }
  }
  __syncthreads();
  if (tid < ROWS)
    att[((size_t)b * MF + m) * SN + s0 + tid] = sPartial[tid];
}

// ---------------------------------------------------------------------------
// Kernel C1: masked softmax over S for each (b, m), in place on att
// ---------------------------------------------------------------------------
__global__ void k_softmax(float* __restrict__ att, const unsigned char* __restrict__ mask) {
  __shared__ float red[256];
  const int m = blockIdx.x, b = blockIdx.y, tid = threadIdx.x;
  float* row = att + ((size_t)b * MF + m) * SN;
  const unsigned char* mk = mask + (size_t)b * SN;
  float x[4];
#pragma unroll
  for (int j = 0; j < 4; ++j) {
    int s = tid + 256 * j;
    x[j] = mk[s] ? -INFINITY : row[s];
  }
  float lm = fmaxf(fmaxf(x[0], x[1]), fmaxf(x[2], x[3]));
  red[tid] = lm; __syncthreads();
  for (int off = 128; off > 0; off >>= 1) {
    if (tid < off) red[tid] = fmaxf(red[tid], red[tid + off]);
    __syncthreads();
  }
  float mx = red[0]; __syncthreads();
  float e[4], ls = 0.f;
#pragma unroll
  for (int j = 0; j < 4; ++j) { e[j] = __expf(x[j] - mx); ls += e[j]; }
  red[tid] = ls; __syncthreads();
  for (int off = 128; off > 0; off >>= 1) {
    if (tid < off) red[tid] += red[tid + off];
    __syncthreads();
  }
  float inv = 1.f / red[0];
#pragma unroll
  for (int j = 0; j < 4; ++j) row[tid + 256 * j] = e[j] * inv;
}

// ---------------------------------------------------------------------------
// Kernel C2: max over m, then sent[b,d] = sum_s att_max[b,s] * enc[b,s,d]
// ---------------------------------------------------------------------------
__global__ void k_combine(const float* __restrict__ att, const float* __restrict__ enc,
                          float* __restrict__ out) {
  __shared__ float wv[SN];
  const int b = blockIdx.x, tid = threadIdx.x;
  const float* ab = att + (size_t)b * MF * SN;
  for (int s = tid; s < SN; s += 256) {
    float v = ab[s];
    v = fmaxf(v, ab[SN + s]);
    v = fmaxf(v, ab[2 * SN + s]);
    v = fmaxf(v, ab[3 * SN + s]);
    wv[s] = v;
  }
  __syncthreads();
  float acc[4] = {0.f, 0.f, 0.f, 0.f};
  const float* eb = enc + (size_t)b * SN * DIN;
  for (int s = 0; s < SN; ++s) {
    float wt = wv[s];
    const float* er = eb + (size_t)s * DIN;
#pragma unroll
    for (int j = 0; j < 4; ++j) acc[j] += wt * er[tid + 256 * j];
  }
#pragma unroll
  for (int j = 0; j < 4; ++j) out[(size_t)b * DIN + tid + 256 * j] = acc[j];
}

// ---------------------------------------------------------------------------
extern "C" void kernel_launch(void* const* d_in, const int* in_sizes, int n_in,
                              void* d_out, int out_size, void* d_ws, size_t ws_size,
                              hipStream_t stream) {
  const float*         enc  = (const float*)d_in[0];
  const unsigned char* mask = (const unsigned char*)d_in[1];  // jax bool -> 1 byte
  const float*         a1   = (const float*)d_in[2];
  const float*         a2   = (const float*)d_in[3];
  const float*         Ws   = (const float*)d_in[4];
  const float*         bs   = (const float*)d_in[5];
  const float*         Ww   = (const float*)d_in[6];
  const float*         bw   = (const float*)d_in[7];
  float*               out  = (float*)d_out;

  // Workspace layout (~8.6 MB): all fully overwritten each call.
  char* ws = (char*)d_ws;
  float*          argW = (float*)ws;                         // 32*1024 f32   = 131072 B
  float*          att  = (float*)(ws + 131072);              // 32*4*1024 f32 = 524288 B
  unsigned short* Wt   = (unsigned short*)(ws + 655360);     // 4*1024*1024 bf16 = 8 MB

  // 1) transpose+convert weights for WMMA B-layout (contiguous-K rows)
  k_wt<<<dim3(DIN / 32, DOUT / 32, MF), 256, 0, stream>>>(Ws, Wt);
  // 2) query vector arg[b,o]
  k_arg<<<dim3(BN), 256, 0, stream>>>(a1, a2, Ww, bw, argW);
  // 3) main fused bf16-WMMA GEMM + tanh + contraction with arg
  size_t lds = (size_t)ROWS * LDA * 2 + (size_t)DOUT * 4 * 2 + (size_t)ROWS * 4;  // 140544 B
  hipFuncSetAttribute((const void*)k_att, hipFuncAttributeMaxDynamicSharedMemorySize, (int)lds);
  k_att<<<dim3(SN / ROWS, MF, BN), 256, lds, stream>>>(enc, Wt, bs, argW, att);
  // 4) masked softmax per (b,m)
  k_softmax<<<dim3(MF, BN), 256, 0, stream>>>(att, mask);
  // 5) max over factors + weighted sum of encoder states
  k_combine<<<dim3(BN), 256, 0, stream>>>(att, enc, out);
}
